// GPTLanguageModel_3874060501302
// MI455X (gfx1250) — matrix operand order repaired and checked by reading, not compiled, not run on hardware
//
#include <hip/hip_runtime.h>
#include <cstdint>
#include <cstddef>

// ---------------------------------------------------------------------------
// CDNA5 (gfx1250) GPT forward. All GEMMs run through v_wmma_f32_16x16x32_bf16
// (fp32 accumulate), fp32 tensors in HBM, on-the-fly bf16 conversion in LDS.
// wave32; WMMA requires EXEC all-1s -> all GEMM shapes tile exactly.
// Software-pipelined K-loop: global loads for tile k+1 staged in VGPRs while
// the matrix pipe consumes tile k from LDS; global_prefetch for tile k+2.
// LDS fill uses packed bf16 pairs (ds_store_b32) via native fptrunc->bf16.
// ---------------------------------------------------------------------------

typedef __attribute__((ext_vector_type(16))) __bf16 v16bf;
typedef __attribute__((ext_vector_type(8)))  float  v8f;

union Frag16 { v16bf v; uint32_t u[8]; };

__device__ __forceinline__ uint32_t pack_bf2(float lo, float hi) {
  // fptrunc float->bf16 is round-to-nearest-even; lowers to the native
  // cvt instruction on gfx1250 (software sequence only as fallback).
  union { __bf16 h[2]; uint32_t u; } p;
  p.h[0] = (__bf16)lo;
  p.h[1] = (__bf16)hi;
  return p.u;
}

// ---------------------------------------------------------------------------
// GEMM: C[M,N] = act( A[M,K] @ B[K,N] [+ bias] [+ residual] )
// A, B, C, residual fp32 row-major. Requires M%128==0, N%128==0, K%32==0.
// Block = 256 threads (8 waves), tile 128x128, K-step 32.
// Wave grid 4x2: each wave owns a 32x64 output patch = 2x4 WMMA tiles.
// Fusion pattern is a compile-time template -> branch-free epilogue.
// ---------------------------------------------------------------------------
template <int HAS_BIAS, int HAS_RES, int DO_RELU>
__global__ __launch_bounds__(256) void gemm_bf16_wmma(
    const float* __restrict__ A, const float* __restrict__ B,
    float* __restrict__ C, const float* __restrict__ bias,
    const float* __restrict__ residual, int M, int N, int K) {

  __shared__ unsigned short lA[128][34];  // [m][k], +2 pad
  __shared__ unsigned short lB[128][34];  // [n][k] (transposed), +2 pad

  const int tid  = threadIdx.x;
  const int lane = tid & 31;
  const int wave = tid >> 5;
  const int wm   = wave >> 1;   // 0..3 -> 32-row band
  const int wn   = wave & 1;    // 0..1 -> 64-col band
  const int half = lane >> 4;   // lane group (0: lanes 0-15, 1: 16-31)
  const int l16  = lane & 15;

  const int blockM = blockIdx.y * 128;
  const int blockN = blockIdx.x * 128;

  v8f acc[2][4] = {};

  // ---- register staging for the software pipeline (bf16-pair granularity) --
  float2 ra[8];            // A: 8 pairs of K-adjacent elements
  float  rb0[8], rb1[8];   // B: 8 pairs of K-adjacent elements (stride-N apart)

  auto load_tiles = [&](int k0) {
#pragma unroll
    for (int i = 0; i < 8; ++i) {
      int p = i * 256 + tid;                // pair index 0..2047
      int r = p >> 4, c = (p & 15) * 2;     // A tile: 128 rows x 16 K-pairs
      ra[i] = *(const float2*)&A[(size_t)(blockM + r) * K + (k0 + c)];
      int nc = p & 127, kr = (p >> 7) * 2;  // B tile: 16 K-pairs x 128 cols
      rb0[i] = B[(size_t)(k0 + kr)     * N + (blockN + nc)];
      rb1[i] = B[(size_t)(k0 + kr + 1) * N + (blockN + nc)];
    }
  };
  auto store_tiles = [&]() {
#pragma unroll
    for (int i = 0; i < 8; ++i) {
      int p = i * 256 + tid;
      int r = p >> 4, c = (p & 15) * 2;
      *(uint32_t*)&lA[r][c] = pack_bf2(ra[i].x, ra[i].y);
      int nc = p & 127, kr = (p >> 7) * 2;
      *(uint32_t*)&lB[nc][kr] = pack_bf2(rb0[i], rb1[i]);
    }
  };

  // prologue: tile 0 -> regs -> LDS
  load_tiles(0);
  store_tiles();
  __syncthreads();

  for (int k0 = 0; k0 < K; k0 += 32) {
    const bool nxt = (k0 + 32) < K;
    if (nxt) {
      load_tiles(k0 + 32);  // global loads in flight during WMMA below
      if (k0 + 64 < K) {
        // pull tile k+2 toward the caches (gfx1250 global_prefetch path)
        int r = tid >> 5, c = tid & 31;
        __builtin_prefetch(&A[(size_t)(blockM + r) * K + (k0 + 64 + c)], 0, 3);
        int kr = tid >> 7, nc = tid & 127;
        __builtin_prefetch(&B[(size_t)(k0 + 64 + kr) * N + (blockN + nc)], 0, 3);
      }
    }

    // ---- A fragments (ISA 7.12.2, 16-bit A 16x32):
    // lanes 0-15: VGPR0-3 K=0..7, VGPR4-7 K=16..23; lanes 16-31: +8
    Frag16 af[2];
#pragma unroll
    for (int mi = 0; mi < 2; ++mi) {
      int row = wm * 32 + mi * 16 + l16;
#pragma unroll
      for (int r = 0; r < 8; ++r) {
        int kb = (r < 4) ? (2 * r + 8 * half) : (16 + 2 * (r - 4) + 8 * half);
        af[mi].u[r] = *(const uint32_t*)&lA[row][kb];
      }
    }
    // ---- B fragments: same K order as A (e<8: K=e+8*half, e>=8: K=16+8*half+e-8), N = lane&15
    Frag16 bfg[4];
#pragma unroll
    for (int ni = 0; ni < 4; ++ni) {
      int col = wn * 64 + ni * 16 + l16;
#pragma unroll
      for (int r = 0; r < 8; ++r) {
        int kb = (r < 4) ? (2 * r + 8 * half) : (16 + 2 * (r - 4) + 8 * half);
        bfg[ni].u[r] = *(const uint32_t*)&lB[col][kb];
      }
    }

    __syncthreads();  // all LDS reads of this tile done (frags live in regs)

#pragma unroll
    for (int mi = 0; mi < 2; ++mi)
#pragma unroll
      for (int ni = 0; ni < 4; ++ni)
        acc[mi][ni] = __builtin_amdgcn_wmma_f32_16x16x32_bf16(
            false, af[mi].v, false, bfg[ni].v, (short)0, acc[mi][ni],
            false, false);

    if (nxt) store_tiles();  // DS writes overlap matrix pipe
    __syncthreads();
  }

  // ---- Epilogue (C/D layout: VGPR r -> M = r + 8*half, N = lane&15) ----
#pragma unroll
  for (int ni = 0; ni < 4; ++ni) {
    const int nCol = blockN + wn * 64 + ni * 16 + l16;
    const float bv = HAS_BIAS ? bias[nCol] : 0.f;
#pragma unroll
    for (int mi = 0; mi < 2; ++mi) {
      const int mBase = blockM + wm * 32 + mi * 16 + half * 8;
#pragma unroll
      for (int r = 0; r < 8; ++r) {
        size_t off = (size_t)(mBase + r) * N + nCol;
        float val = acc[mi][ni][r] + bv;
        if (HAS_RES) val += residual[off];
        if (DO_RELU) val = fmaxf(val, 0.f);
        C[off] = val;
      }
    }
  }
}

// ---------------------------------------------------------------------------
// Embedding: x[b,t,:] = tok_emb[idx[b,t],:] + pos_emb[t,:]
// ---------------------------------------------------------------------------
__global__ __launch_bounds__(256) void embed_kernel(
    const int* __restrict__ idx, const float* __restrict__ tok,
    const float* __restrict__ pos, float* __restrict__ x,
    int T, int E, int total) {
  int gid = blockIdx.x * 256 + threadIdx.x;
  if (gid >= total) return;
  int e  = gid % E;
  int bt = gid / E;
  int t  = bt % T;
  x[gid] = tok[(size_t)idx[bt] * E + e] + pos[(size_t)t * E + e];
}

// ---------------------------------------------------------------------------
// LayerNorm: one 256-thread block per row of E=768 elements.
// ---------------------------------------------------------------------------
__global__ __launch_bounds__(256) void layernorm_kernel(
    const float* __restrict__ x, const float* __restrict__ g,
    const float* __restrict__ b, float* __restrict__ y, int E) {
  __shared__ float s1[256];
  __shared__ float s2[256];
  const int row = blockIdx.x;
  const int tid = threadIdx.x;
  const float* xr = x + (size_t)row * E;
  float sum = 0.f, sq = 0.f;
  for (int e = tid; e < E; e += 256) {
    float v = xr[e];
    sum += v; sq += v * v;
  }
  s1[tid] = sum; s2[tid] = sq;
  __syncthreads();
  for (int s = 128; s > 0; s >>= 1) {
    if (tid < s) { s1[tid] += s1[tid + s]; s2[tid] += s2[tid + s]; }
    __syncthreads();
  }
  const float mean = s1[0] / (float)E;
  const float var  = s2[0] / (float)E - mean * mean;
  const float inv  = rsqrtf(var + 1e-5f);
  float* yr = y + (size_t)row * E;
  for (int e = tid; e < E; e += 256)
    yr[e] = (xr[e] - mean) * inv * g[e] + b[e];
}

// ---------------------------------------------------------------------------
// Causal attention with online softmax. One thread per (b,h,t) query.
// q,k,v,out are [B*T, E] with head h at columns h*64..h*64+63.
// scale = E^-0.5 (matches reference, which scales by n_embd^-0.5).
// ---------------------------------------------------------------------------
__global__ __launch_bounds__(256) void attention_kernel(
    const float* __restrict__ q, const float* __restrict__ k,
    const float* __restrict__ v, float* __restrict__ out,
    int T, int H, int E, float scale) {
  const int gid = blockIdx.x * 256 + threadIdx.x;
  const int t = gid % T;
  const int h = (gid / T) % H;
  const int b = gid / (T * H);
  const int D = E / H;  // 64

  const size_t qoff = ((size_t)(b * T + t)) * E + h * D;
  float qv[64];
#pragma unroll
  for (int d = 0; d < 64; ++d) qv[d] = q[qoff + d];

  float m = -3.0e38f, l = 0.f;
  float acc[64];
#pragma unroll
  for (int d = 0; d < 64; ++d) acc[d] = 0.f;

  for (int s = 0; s <= t; ++s) {
    const size_t koff = ((size_t)(b * T + s)) * E + h * D;
    const float* kp = k + koff;
    float dot = 0.f;
#pragma unroll
    for (int d = 0; d < 64; ++d) dot += qv[d] * kp[d];
    dot *= scale;
    float nm   = fmaxf(m, dot);
    float corr = __expf(m - nm);
    float p    = __expf(dot - nm);
    l = l * corr + p;
    const float* vp = v + koff;
#pragma unroll
    for (int d = 0; d < 64; ++d) acc[d] = acc[d] * corr + p * vp[d];
    m = nm;
  }
  const float inv = 1.f / l;
#pragma unroll
  for (int d = 0; d < 64; ++d) out[qoff + d] = acc[d] * inv;
}

// ---------------------------------------------------------------------------
// Host-side orchestration.
// ---------------------------------------------------------------------------
static inline void launch_gemm(const float* A, const float* B, float* C,
                               const float* bias, const float* residual,
                               int M, int N, int K, int relu,
                               hipStream_t stream) {
  dim3 grid(N / 128, M / 128);
  if (bias && residual)
    gemm_bf16_wmma<1, 1, 0><<<grid, 256, 0, stream>>>(A, B, C, bias, residual, M, N, K);
  else if (bias && relu)
    gemm_bf16_wmma<1, 0, 1><<<grid, 256, 0, stream>>>(A, B, C, bias, residual, M, N, K);
  else if (bias)
    gemm_bf16_wmma<1, 0, 0><<<grid, 256, 0, stream>>>(A, B, C, bias, residual, M, N, K);
  else
    gemm_bf16_wmma<0, 0, 0><<<grid, 256, 0, stream>>>(A, B, C, bias, residual, M, N, K);
}

extern "C" void kernel_launch(void* const* d_in, const int* in_sizes, int n_in,
                              void* d_out, int out_size, void* d_ws, size_t ws_size,
                              hipStream_t stream) {
  (void)in_sizes; (void)n_in; (void)out_size; (void)ws_size;

  constexpr int Lc = 6, Ec = 768, Hc = 12, Bc = 4, Tc = 1024, Vc = 32000;
  constexpr int MT = Bc * Tc;        // 4096 rows
  constexpr int E4 = 4 * Ec;         // 3072

  const int*   idx     = (const int*)  d_in[0];
  const float* tok_emb = (const float*)d_in[1];
  const float* pos_emb = (const float*)d_in[2];
  const float* Wq      = (const float*)d_in[3];
  const float* Wk      = (const float*)d_in[4];
  const float* Wv      = (const float*)d_in[5];
  const float* Wo      = (const float*)d_in[6];
  const float* bo      = (const float*)d_in[7];
  const float* ln1_g   = (const float*)d_in[8];
  const float* ln1_b   = (const float*)d_in[9];
  const float* ln2_g   = (const float*)d_in[10];
  const float* ln2_b   = (const float*)d_in[11];
  const float* W1      = (const float*)d_in[12];
  const float* b1      = (const float*)d_in[13];
  const float* W2      = (const float*)d_in[14];
  const float* b2      = (const float*)d_in[15];
  const float* lnf_g   = (const float*)d_in[16];
  const float* lnf_b   = (const float*)d_in[17];
  const float* Whead   = (const float*)d_in[18];
  const float* bhead   = (const float*)d_in[19];
  float* logits        = (float*)d_out;

  // Workspace layout (floats). Q,K,V,ATT are contiguous (MT*4E total) so the
  // MLP hidden buffer M1 aliases them once attention is consumed.
  float* ws  = (float*)d_ws;
  float* X   = ws;                     // MT*E : residual stream
  float* Hh  = X   + (size_t)MT * Ec;  // MT*E : layernorm output
  float* Q   = Hh  + (size_t)MT * Ec;  // MT*E
  float* Kb  = Q   + (size_t)MT * Ec;  // MT*E
  float* Vb  = Kb  + (size_t)MT * Ec;  // MT*E
  float* ATT = Vb  + (size_t)MT * Ec;  // MT*E
  float* XA  = ATT + (size_t)MT * Ec;  // MT*E : post-attention residual
  float* M1  = Q;                      // MT*4E: aliases Q..ATT (dead by then)

  const float scale = 0.03608439182435161f;  // 768^-0.5 (E^-0.5 per reference)

  // x = tok_emb[idx] + pos_emb
  {
    int total = MT * Ec;
    embed_kernel<<<(total + 255) / 256, 256, 0, stream>>>(
        idx, tok_emb, pos_emb, X, Tc, Ec, total);
  }

  for (int i = 0; i < Lc; ++i) {
    const size_t wEE = (size_t)i * Ec * Ec;
    const size_t wE4 = (size_t)i * Ec * E4;
    const size_t vE  = (size_t)i * Ec;
    const size_t v4  = (size_t)i * E4;

    // h = LN1(x)
    layernorm_kernel<<<MT, 256, 0, stream>>>(X, ln1_g + vE, ln1_b + vE, Hh, Ec);
    // q,k,v = h @ Wq/Wk/Wv
    launch_gemm(Hh, Wq + wEE, Q,  nullptr, nullptr, MT, Ec, Ec, 0, stream);
    launch_gemm(Hh, Wk + wEE, Kb, nullptr, nullptr, MT, Ec, Ec, 0, stream);
    launch_gemm(Hh, Wv + wEE, Vb, nullptr, nullptr, MT, Ec, Ec, 0, stream);
    // attn = causal-softmax(q k^T * scale) v
    attention_kernel<<<(Bc * Hc * Tc) / 256, 256, 0, stream>>>(
        Q, Kb, Vb, ATT, Tc, Hc, Ec, scale);
    // xa = x + attn @ Wo + bo
    launch_gemm(ATT, Wo + wEE, XA, bo + vE, X, MT, Ec, Ec, 0, stream);
    // h = LN2(xa)
    layernorm_kernel<<<MT, 256, 0, stream>>>(XA, ln2_g + vE, ln2_b + vE, Hh, Ec);
    // m1 = relu(h @ W1 + b1)   (m1 aliases Q..ATT; attn already consumed)
    launch_gemm(Hh, W1 + wE4, M1, b1 + v4, nullptr, MT, E4, Ec, 1, stream);
    // x = xa + m1 @ W2 + b2
    launch_gemm(M1, W2 + wE4, X, b2 + vE, XA, MT, Ec, E4, 0, stream);
  }

  // h = LN_f(x); logits = h @ Whead + bhead
  layernorm_kernel<<<MT, 256, 0, stream>>>(X, lnf_g, lnf_b, Hh, Ec);
  launch_gemm(Hh, Whead, logits, bhead, nullptr, MT, Vc, Ec, 0, stream);
}
